// ObjectModule_46136538694233
// MI455X (gfx1250) — 8-bit matrix operand order repaired and checked by reading, compiled, not run on hardware
//
#include <hip/hip_runtime.h>

typedef int v8i __attribute__((ext_vector_type(8)));

#define T_TOTAL   4194304
#define N_DIM     131072
#define CHUNK     8192
#define NBLK      (T_TOTAL / CHUNK)
#define WAVE_TILE 1024
#define NEG_IDLE  (-100000000.0f)
#define NEG_FILL  (-1000.0f)

__device__ __forceinline__ int pack4(int4 v) {
  return v.x | (v.y << 8) | (v.z << 16) | (v.w << 24);
}

__global__ __launch_bounds__(256) void count_kernel(const int* __restrict__ ms,
                                                    const int* __restrict__ ma,
                                                    int* __restrict__ cnt_s,
                                                    int* __restrict__ cnt_a) {
  __shared__ int red[256];
  const int tid = threadIdx.x;
  const long long base4 = (long long)blockIdx.x * (CHUNK / 4);
  const int4* ms4 = (const int4*)ms;
  const int4* ma4 = (const int4*)ma;
  int ss = 0, sa = 0;
#pragma unroll
  for (int i = 0; i < 8; ++i) {
    int4 v = ms4[base4 + tid + i * 256];
    ss += v.x + v.y + v.z + v.w;
    int4 w = ma4[base4 + tid + i * 256];
    sa += w.x + w.y + w.z + w.w;
  }
  red[tid] = ss;
  __syncthreads();
  for (int s = 128; s > 0; s >>= 1) {
    if (tid < s) red[tid] += red[tid + s];
    __syncthreads();
  }
  if (tid == 0) cnt_s[blockIdx.x] = red[0];
  __syncthreads();
  red[tid] = sa;
  __syncthreads();
  for (int s = 128; s > 0; s >>= 1) {
    if (tid < s) red[tid] += red[tid + s];
    __syncthreads();
  }
  if (tid == 0) cnt_a[blockIdx.x] = red[0];
}

__global__ __launch_bounds__(NBLK) void scan_kernel(int* __restrict__ cnt_s,
                                                    int* __restrict__ cnt_a) {
  __shared__ int buf[NBLK];
  const int tid = threadIdx.x;
  int v = cnt_s[tid];
  buf[tid] = v;
  __syncthreads();
  for (int off = 1; off < NBLK; off <<= 1) {
    int t = (tid >= off) ? buf[tid - off] : 0;
    __syncthreads();
    buf[tid] += t;
    __syncthreads();
  }
  cnt_s[tid] = buf[tid] - v;
  __syncthreads();
  v = cnt_a[tid];
  buf[tid] = v;
  __syncthreads();
  for (int off = 1; off < NBLK; off <<= 1) {
    int t = (tid >= off) ? buf[tid - off] : 0;
    __syncthreads();
    buf[tid] += t;
    __syncthreads();
  }
  cnt_a[tid] = buf[tid] - v;
}

__global__ __launch_bounds__(256) void expand_kernel(
    const float* __restrict__ rgb, const float* __restrict__ ls,
    const float* __restrict__ la, const int* __restrict__ ms,
    const int* __restrict__ ma, const float* __restrict__ idle_states,
    const int* __restrict__ off_s, const int* __restrict__ off_a,
    float* __restrict__ out_rgb, float* __restrict__ out_ls,
    float* __restrict__ out_la) {
  __shared__ int ldsRow[8][16];
  __shared__ int ldsWave[8];

  const int tid  = threadIdx.x;
  const int wv   = tid >> 5;
  const int lane = tid & 31;
  const int half = lane >> 4;
  const int l15  = lane & 15;

  const int chunkBase = blockIdx.x * CHUNK;
  const int tileBase  = chunkBase + wv * WAVE_TILE;
  const int bk        = blockIdx.x >> 4;
  const float idle    = idle_states[bk];
  const float scale   = 1.0f - idle;
  const float idleTerm = idle * NEG_IDLE;

  __builtin_prefetch(ma + tileBase + lane * 32, 0, 1);

  v8i bpat;
  {
    const int lim = 4 * (l15 + 1);
#pragma unroll
    for (int r = 0; r < 8; ++r) {
      const int kb = ((r & 1) << 2) + (((r >> 1) & 1) << 4) + ((r >> 2) << 5) + (half << 3);
      bpat[r] = (kb < lim) ? 0x01010101 : 0;
    }
  }
  const v8i czero = {};

  v8i aS;
  {
    const int4* mp = (const int4*)(ms + tileBase);
#pragma unroll
    for (int r = 0; r < 8; ++r) {
      const int kb = ((r & 1) << 2) + (((r >> 1) & 1) << 4) + ((r >> 2) << 5) + (half << 3);
      aS[r] = pack4(mp[l15 * 16 + (kb >> 2)]);
    }
  }
  v8i dS = __builtin_amdgcn_wmma_i32_16x16x64_iu8(false, aS, false, bpat, czero, false, false);

  if (l15 == 15) {
#pragma unroll
    for (int v = 0; v < 8; ++v) ldsRow[wv][half * 8 + v] = dS[v];
  }
  __syncthreads();
  int exclS[16];
  int baseS;
  {
    int run = 0;
#pragma unroll
    for (int m = 0; m < 16; ++m) { exclS[m] = run; run += ldsRow[wv][m]; }
    if (lane == 0) ldsWave[wv] = run;
  }
  __syncthreads();
  baseS = off_s[blockIdx.x];
#pragma unroll
  for (int w = 0; w < 8; ++w)
    if (w < wv) baseS += ldsWave[w];

#pragma unroll
  for (int v = 0; v < 8; ++v) {
    const int M        = v + (half << 3);
    const int elemBase = tileBase + M * 64 + (l15 << 2);
    const int4 mv      = ((const int4*)ms)[elemBase >> 2];
    const int g        = mv.x + mv.y + mv.z + mv.w;
    const int exM      = half ? exclS[8 + v] : exclS[v];
    int cnt            = baseS + exM + dS[v] - g;
    int mk[4] = {mv.x, mv.y, mv.z, mv.w};
    float lo[4], rr[12];
#pragma unroll
    for (int j = 0; j < 4; ++j) {
      if (mk[j] != 0) {
        lo[j] = ls[cnt] * scale + idleTerm;
        const float* rp = rgb + cnt * 3;
        rr[3 * j + 0] = rp[0] * scale;
        rr[3 * j + 1] = rp[1] * scale;
        rr[3 * j + 2] = rp[2] * scale;
        ++cnt;
      } else {
        lo[j] = NEG_FILL;
        rr[3 * j + 0] = 0.0f; rr[3 * j + 1] = 0.0f; rr[3 * j + 2] = 0.0f;
      }
    }
    ((float4*)out_ls)[elemBase >> 2] = make_float4(lo[0], lo[1], lo[2], lo[3]);
    float4* orp = (float4*)(out_rgb + (size_t)elemBase * 3);
    orp[0] = make_float4(rr[0], rr[1], rr[2],  rr[3]);
    orp[1] = make_float4(rr[4], rr[5], rr[6],  rr[7]);
    orp[2] = make_float4(rr[8], rr[9], rr[10], rr[11]);
  }

  __syncthreads();
  v8i aA;
  {
    const int4* mp = (const int4*)(ma + tileBase);
#pragma unroll
    for (int r = 0; r < 8; ++r) {
      const int kb = ((r & 1) << 2) + (((r >> 1) & 1) << 4) + ((r >> 2) << 5) + (half << 3);
      aA[r] = pack4(mp[l15 * 16 + (kb >> 2)]);
    }
  }
  v8i dA = __builtin_amdgcn_wmma_i32_16x16x64_iu8(false, aA, false, bpat, czero, false, false);

  if (l15 == 15) {
#pragma unroll
    for (int v = 0; v < 8; ++v) ldsRow[wv][half * 8 + v] = dA[v];
  }
  __syncthreads();
  int exclA[16];
  int baseA;
  {
    int run = 0;
#pragma unroll
    for (int m = 0; m < 16; ++m) { exclA[m] = run; run += ldsRow[wv][m]; }
    if (lane == 0) ldsWave[wv] = run;
  }
  __syncthreads();
  baseA = off_a[blockIdx.x];
#pragma unroll
  for (int w = 0; w < 8; ++w)
    if (w < wv) baseA += ldsWave[w];

#pragma unroll
  for (int v = 0; v < 8; ++v) {
    const int M        = v + (half << 3);
    const int elemBase = tileBase + M * 64 + (l15 << 2);
    const int4 mv      = ((const int4*)ma)[elemBase >> 2];
    const int g        = mv.x + mv.y + mv.z + mv.w;
    const int exM      = half ? exclA[8 + v] : exclA[v];
    int cnt            = baseA + exM + dA[v] - g;
    int mk[4] = {mv.x, mv.y, mv.z, mv.w};
    float lo[4];
#pragma unroll
    for (int j = 0; j < 4; ++j) {
      if (mk[j] != 0) { lo[j] = la[cnt] * scale + idleTerm; ++cnt; }
      else           { lo[j] = NEG_FILL; }
    }
    ((float4*)out_la)[elemBase >> 2] = make_float4(lo[0], lo[1], lo[2], lo[3]);
  }
}

extern "C" void kernel_launch(void* const* d_in, const int* in_sizes, int n_in,
                              void* d_out, int out_size, void* d_ws, size_t ws_size,
                              hipStream_t stream) {
  const float* rgb   = (const float*)d_in[0];
  const float* ls    = (const float*)d_in[1];
  const float* la    = (const float*)d_in[2];
  const int*   msk_s = (const int*)d_in[3];
  const int*   msk_a = (const int*)d_in[4];
  const float* idle  = (const float*)d_in[5];

  float* out     = (float*)d_out;
  float* out_rgb = out;
  float* out_ls  = out + (size_t)3 * T_TOTAL;
  float* out_la  = out + (size_t)4 * T_TOTAL;

  int* cnt_s = (int*)d_ws;
  int* cnt_a = cnt_s + NBLK;

  count_kernel<<<NBLK, 256, 0, stream>>>(msk_s, msk_a, cnt_s, cnt_a);
  scan_kernel<<<1, NBLK, 0, stream>>>(cnt_s, cnt_a);
  expand_kernel<<<NBLK, 256, 0, stream>>>(rgb, ls, la, msk_s, msk_a, idle,
                                          cnt_s, cnt_a, out_rgb, out_ls, out_la);
}
